// TransMIL_52089363366133
// MI455X (gfx1250) — matrix operand order repaired and checked by reading, not compiled, not run on hardware
//
#include <hip/hip_runtime.h>
#include <hip/hip_bf16.h>

typedef __bf16 bf16;
typedef __attribute__((ext_vector_type(16))) __bf16 v16bf;
typedef __attribute__((ext_vector_type(8)))  __bf16 v8bf;
typedef __attribute__((ext_vector_type(8)))  float  v8f;

#define THREADS 256
#define BM 128
#define BN 64
#define BK 64          // two 16x16x32 WMMA sub-steps per LDS stage
#define SA (BK + 8)    // 72 bf16 = 144B row stride (16B aligned, conflict-free)
#define SBS (BK + 8)

#define NP    16640   // padded sequence (255 zero rows + 16385 tokens)
#define NTOK  16385
#define DMODEL 512
#define NHEAD 8
#define DHEAD 64
#define NLM   256
#define LSEG  65
#define PADR  255

// ---------------- WMMA fragment loaders (ISA 7.12.2 layouts) ----------------
__device__ __forceinline__ v16bf frag_a(const bf16* sA, int lane, int row0, int ko) {
  int l = lane & 15, hi = lane >> 4;
  const bf16* p = sA + (size_t)(row0 + l) * SA + ko;
  union { v16bf v; v8bf h[2]; } u;
  u.h[0] = *(const v8bf*)(p + (hi ? 8 : 0));
  u.h[1] = *(const v8bf*)(p + (hi ? 24 : 16));
  return u.v;
}
__device__ __forceinline__ v16bf frag_b(const bf16* sB, int lane, int n0, int ko) {
  int l = lane & 15, hi = lane >> 4;
  const bf16* p = sB + (size_t)(n0 + l) * SBS + ko;
  union { v16bf v; v8bf h[2]; } u;
  u.h[0] = *(const v8bf*)(p + (hi ? 8 : 0));
  u.h[1] = *(const v8bf*)(p + (hi ? 24 : 16));
  return u.v;
}

// ---------------- Generic batched WMMA GEMM (double-buffered) ----------------
// C[b] = alpha * A[b] @ op(B[b]) (+bias) (+relu)
// BNK=1: B given row-major [N,K] (i.e., B^T); BNK=0: B row-major [K,N].
// EPI: 0 none, 1 +bias, 2 +bias+relu. Requires M%128==0, N%64==0, K%64==0.
template<int EPI, int BNK>
__global__ __launch_bounds__(THREADS)
void gemm_kernel(const float* __restrict__ A, const float* __restrict__ B,
                 float* __restrict__ C, const float* __restrict__ bias,
                 int M, int N, int K, int lda, int ldb, int ldc,
                 long long sAb, long long sBb, long long sCb, float alpha)
{
  __shared__ bf16 sA[2][BM * SA];
  __shared__ bf16 sB[2][BN * SBS];
  int tid  = threadIdx.x;
  int lane = tid & 31, wave = tid >> 5;
  int wm = wave >> 1, wn = wave & 1;

  const float* Ab = A + (long long)blockIdx.z * sAb + (size_t)(blockIdx.x * BM) * lda;
  const float* Bb = B + (long long)blockIdx.z * sBb;
  float*       Cb = C + (long long)blockIdx.z * sCb + (size_t)(blockIdx.x * BM) * ldc
                      + (size_t)blockIdx.y * BN;

  // Register staging for the software pipeline (global -> reg -> LDS)
  float4 ra[8];                 // A tile: 128x64 f32 = 2048 float4 / 256 thr = 8
  float4 rb[4];                 // B tile (BNK=1): 64x64 f32 = 1024 float4 / 256 = 4
  float  rbs[16];               // B tile (BNK=0): 4096 scalars / 256 = 16

  auto gloadA = [&](int kt) {
    const float* Ag = Ab + kt * BK;
    #pragma unroll
    for (int j = 0; j < 8; ++j) {
      int idx = tid + j * THREADS;
      int r = idx >> 4, c4 = (idx & 15) << 2;
      ra[j] = *(const float4*)(Ag + (size_t)r * lda + c4);
    }
  };
  auto storeA = [&](bf16* dst) {
    #pragma unroll
    for (int j = 0; j < 8; ++j) {
      int idx = tid + j * THREADS;
      int r = idx >> 4, c4 = (idx & 15) << 2;
      bf16* d = dst + r * SA + c4;
      d[0] = (bf16)ra[j].x; d[1] = (bf16)ra[j].y; d[2] = (bf16)ra[j].z; d[3] = (bf16)ra[j].w;
    }
  };
  auto gloadB = [&](int kt) {
    if (BNK) {
      const float* Bg = Bb + (size_t)(blockIdx.y * BN) * ldb + kt * BK;
      #pragma unroll
      for (int j = 0; j < 4; ++j) {
        int idx = tid + j * THREADS;
        int r = idx >> 4, c4 = (idx & 15) << 2;
        rb[j] = *(const float4*)(Bg + (size_t)r * ldb + c4);
      }
    } else {
      const float* Bg = Bb + (size_t)(kt * BK) * ldb + (size_t)blockIdx.y * BN;
      int n = tid & 63, kb = tid >> 6;        // kb in 0..3
      #pragma unroll
      for (int j = 0; j < 16; ++j)
        rbs[j] = Bg[(size_t)(kb + j * 4) * ldb + n];
    }
  };
  auto storeB = [&](bf16* dst) {
    if (BNK) {
      #pragma unroll
      for (int j = 0; j < 4; ++j) {
        int idx = tid + j * THREADS;
        int r = idx >> 4, c4 = (idx & 15) << 2;
        bf16* d = dst + r * SBS + c4;
        d[0] = (bf16)rb[j].x; d[1] = (bf16)rb[j].y; d[2] = (bf16)rb[j].z; d[3] = (bf16)rb[j].w;
      }
    } else {
      int n = tid & 63, kb = tid >> 6;
      #pragma unroll
      for (int j = 0; j < 16; ++j)
        dst[n * SBS + kb + j * 4] = (bf16)rbs[j];
    }
  };

  v8f acc[2][2] = {};
  int ksteps = K / BK;

  // Prologue: stage 0
  gloadA(0); gloadB(0);
  storeA(sA[0]); storeB(sB[0]);

  for (int kt = 0; kt < ksteps; ++kt) {
    int cur = kt & 1;
    bool more = (kt + 1) < ksteps;
    if (more) {                         // overlap next-stage global loads with WMMA
      gloadA(kt + 1); gloadB(kt + 1);
      __builtin_prefetch(Ab + (kt + 2) * BK + (size_t)(tid >> 1) * lda, 0, 1);
    }
    __syncthreads();                    // stage `cur` visible to all waves
    #pragma unroll
    for (int ks = 0; ks < 2; ++ks) {    // two 16x16x32 sub-steps
      int ko = ks * 32;
      v16bf a0 = frag_a(sA[cur], lane, wm * 32,      ko);
      v16bf a1 = frag_a(sA[cur], lane, wm * 32 + 16, ko);
      v16bf b0 = frag_b(sB[cur], lane, wn * 32,      ko);
      v16bf b1 = frag_b(sB[cur], lane, wn * 32 + 16, ko);
      acc[0][0] = __builtin_amdgcn_wmma_f32_16x16x32_bf16(false, a0, false, b0, (short)0, acc[0][0], false, false);
      acc[0][1] = __builtin_amdgcn_wmma_f32_16x16x32_bf16(false, a0, false, b1, (short)0, acc[0][1], false, false);
      acc[1][0] = __builtin_amdgcn_wmma_f32_16x16x32_bf16(false, a1, false, b0, (short)0, acc[1][0], false, false);
      acc[1][1] = __builtin_amdgcn_wmma_f32_16x16x32_bf16(false, a1, false, b1, (short)0, acc[1][1], false, false);
    }
    if (more) { storeA(sA[cur ^ 1]); storeB(sB[cur ^ 1]); }
  }

  int rbl = (lane >> 4) << 3;   // lanes>=16 hold rows M=8..15 of each tile
  int cl  = lane & 15;
  #pragma unroll
  for (int tm = 0; tm < 2; ++tm)
  #pragma unroll
  for (int tn = 0; tn < 2; ++tn) {
    int r0 = wm * 32 + tm * 16 + rbl;
    int c0 = wn * 32 + tn * 16 + cl;
    float* cp = Cb + (size_t)r0 * ldc + c0;
    int cg = blockIdx.y * BN + c0;
    v8f a = acc[tm][tn];
    #pragma unroll
    for (int r = 0; r < 8; ++r) {
      float v = alpha * a[r];
      if (EPI >= 1) v += bias[cg];
      if (EPI == 2) v = fmaxf(v, 0.f);
      cp[(size_t)r * ldc] = v;
    }
  }
}

// ---------------- Elementwise / reduction kernels ----------------
__global__ void k_fill_zero(float* p, long long n) {
  long long i = (long long)blockIdx.x * blockDim.x + threadIdx.x;
  if (i < n) p[i] = 0.f;
}
__global__ void k_copy(float* dst, const float* src, long long n) {
  long long i = (long long)blockIdx.x * blockDim.x + threadIdx.x;
  if (i < n) dst[i] = src[i];
}
__global__ void k_build_tokens(const float* h0, const float* cls, float* tok, long long n) {
  long long i = (long long)blockIdx.x * blockDim.x + threadIdx.x;
  if (i >= n) return;
  int c = (int)(i & (DMODEL - 1));
  tok[i] = (i < DMODEL) ? cls[c] : h0[i - DMODEL];
}
__global__ void k_res_add(float* tok, const float* outp, long long n) {
  long long i = (long long)blockIdx.x * blockDim.x + threadIdx.x;
  if (i < n) tok[i] += outp[(long long)PADR * DMODEL + i];
}
__global__ void k_id_minus(float* dst, const float* src, float c, long long n) {
  long long idx = (long long)blockIdx.x * blockDim.x + threadIdx.x;
  if (idx >= n) return;
  int i = (int)((idx >> 8) & 255), j = (int)(idx & 255);
  dst[idx] = (i == j ? c : 0.f) - src[idx];
}

__global__ void k_layernorm(const float* __restrict__ src, const float* __restrict__ g,
                            const float* __restrict__ b, float* __restrict__ dst, int cols) {
  __shared__ float red[THREADS];
  int row = blockIdx.x, tid = threadIdx.x;
  const float* x = src + (size_t)row * cols;
  float s = 0.f;
  for (int c = tid; c < cols; c += THREADS) s += x[c];
  red[tid] = s; __syncthreads();
  for (int st = THREADS / 2; st > 0; st >>= 1) { if (tid < st) red[tid] += red[tid + st]; __syncthreads(); }
  float mu = red[0] / cols; __syncthreads();
  float vs = 0.f;
  for (int c = tid; c < cols; c += THREADS) { float d = x[c] - mu; vs += d * d; }
  red[tid] = vs; __syncthreads();
  for (int st = THREADS / 2; st > 0; st >>= 1) { if (tid < st) red[tid] += red[tid + st]; __syncthreads(); }
  float rstd = rsqrtf(red[0] / cols + 1e-5f);
  float* o = dst + (size_t)row * cols;
  for (int c = tid; c < cols; c += THREADS) o[c] = (x[c] - mu) * rstd * g[c] + b[c];
}

__global__ void k_softmax(float* x, int cols) {
  __shared__ float red[THREADS];
  int tid = threadIdx.x;
  float* p = x + (size_t)blockIdx.x * cols;
  float m = -3.4e38f;
  for (int c = tid; c < cols; c += THREADS) m = fmaxf(m, p[c]);
  red[tid] = m; __syncthreads();
  for (int st = THREADS / 2; st > 0; st >>= 1) { if (tid < st) red[tid] = fmaxf(red[tid], red[tid + st]); __syncthreads(); }
  m = red[0]; __syncthreads();
  float s = 0.f;
  for (int c = tid; c < cols; c += THREADS) { float e = __expf(p[c] - m); p[c] = e; s += e; }
  red[tid] = s; __syncthreads();
  for (int st = THREADS / 2; st > 0; st >>= 1) { if (tid < st) red[tid] += red[tid + st]; __syncthreads(); }
  float inv = 1.f / red[0];
  for (int c = tid; c < cols; c += THREADS) p[c] *= inv;
}

__global__ void k_split_qkv(const float* __restrict__ qkv, float* __restrict__ q,
                            float* __restrict__ k, float* __restrict__ v, long long n) {
  long long idx = (long long)blockIdx.x * blockDim.x + threadIdx.x;
  if (idx >= n) return;
  int i = (int)(idx >> 9);        // /512
  int c = (int)(idx & 511);
  int h = c >> 6, d = c & 63;
  size_t dst = ((size_t)h * NP + i) * DHEAD + d;
  size_t src = (size_t)i * (3 * DMODEL) + c;
  q[dst] = qkv[src] * 0.125f;     // DIM_HEAD^-0.5
  k[dst] = qkv[src + DMODEL];
  v[dst] = qkv[src + 2 * DMODEL];
}

__global__ void k_landmark(const float* __restrict__ src, float* __restrict__ dst) {
  int idx = blockIdx.x * blockDim.x + threadIdx.x;   // 8*256*64
  if (idx >= NHEAD * NLM * DHEAD) return;
  int h = idx >> 14, rem = idx & 16383;
  int j = rem >> 6, d = rem & 63;
  const float* p = src + ((size_t)h * NP + (size_t)j * LSEG) * DHEAD + d;
  float s = 0.f;
  for (int r = 0; r < LSEG; ++r) s += p[(size_t)r * DHEAD];
  dst[idx] = s * (1.f / LSEG);
}

__global__ void k_pinv_init(const float* __restrict__ a2, float* __restrict__ z) {
  __shared__ float rs[NLM], cs[NLM];
  int h = blockIdx.x, t = threadIdx.x;               // 256 threads
  const float* A = a2 + (size_t)h * NLM * NLM;
  float*       Z = z  + (size_t)h * NLM * NLM;
  float r = 0.f, c = 0.f;
  for (int j = 0; j < NLM; ++j) { r += fabsf(A[t * NLM + j]); c += fabsf(A[j * NLM + t]); }
  rs[t] = r; cs[t] = c; __syncthreads();
  for (int st = NLM / 2; st > 0; st >>= 1) {
    if (t < st) { rs[t] = fmaxf(rs[t], rs[t + st]); cs[t] = fmaxf(cs[t], cs[t + st]); }
    __syncthreads();
  }
  float scale = 1.f / (rs[0] * cs[0]);
  __syncthreads();
  for (int i = 0; i < NLM; ++i) Z[t * NLM + i] = A[i * NLM + t] * scale;  // x^T/(col*row)
}

// cat[i][h*64+d] = attn_out[h][i][d] + sum_t rw[h][t] * v[h][i+t-16][d]
__global__ void k_cat_conv(const float* __restrict__ outhd, const float* __restrict__ v,
                           const float* __restrict__ rw, float* __restrict__ cat) {
  int i = blockIdx.x;
  for (int c = threadIdx.x; c < DMODEL; c += THREADS) {
    int h = c >> 6, d = c & 63;
    float acc = outhd[((size_t)h * NP + i) * DHEAD + d];
    const float* vh = v + (size_t)h * NP * DHEAD + d;
    #pragma unroll
    for (int t = 0; t < 33; ++t) {
      int j = i + t - 16;
      if (j >= 0 && j < NP) acc += rw[h * 33 + t] * vh[(size_t)j * DHEAD];
    }
    cat[(size_t)i * DMODEL + c] = acc;
  }
}

// PPEG: out = dw7(f)+b7 + f + dw5(f)+b5 + dw3(f)+b3 on 128x128 grid
__global__ void k_ppeg(const float* __restrict__ tok,
                       const float* __restrict__ w7, const float* __restrict__ b7,
                       const float* __restrict__ w5, const float* __restrict__ b5,
                       const float* __restrict__ w3, const float* __restrict__ b3,
                       float* __restrict__ outp) {
  int pos = blockIdx.x;
  int y = pos >> 7, x = pos & 127;
  for (int c = threadIdx.x; c < DMODEL; c += THREADS) {
    float acc = tok[(size_t)(1 + pos) * DMODEL + c] + b7[c] + b5[c] + b3[c];
    for (int dy = -3; dy <= 3; ++dy) {
      int yy = y + dy; if (yy < 0 || yy > 127) continue;
      for (int dx = -3; dx <= 3; ++dx) {
        int xx = x + dx; if (xx < 0 || xx > 127) continue;
        acc += tok[(size_t)(1 + yy * 128 + xx) * DMODEL + c] * w7[c * 49 + (dy + 3) * 7 + (dx + 3)];
      }
    }
    for (int dy = -2; dy <= 2; ++dy) {
      int yy = y + dy; if (yy < 0 || yy > 127) continue;
      for (int dx = -2; dx <= 2; ++dx) {
        int xx = x + dx; if (xx < 0 || xx > 127) continue;
        acc += tok[(size_t)(1 + yy * 128 + xx) * DMODEL + c] * w5[c * 25 + (dy + 2) * 5 + (dx + 2)];
      }
    }
    for (int dy = -1; dy <= 1; ++dy) {
      int yy = y + dy; if (yy < 0 || yy > 127) continue;
      for (int dx = -1; dx <= 1; ++dx) {
        int xx = x + dx; if (xx < 0 || xx > 127) continue;
        acc += tok[(size_t)(1 + yy * 128 + xx) * DMODEL + c] * w3[c * 9 + (dy + 1) * 3 + (dx + 1)];
      }
    }
    outp[(size_t)pos * DMODEL + c] = acc;
  }
}

// Final: LN(token 0) @ fc2_w + fc2_b -> logits, probs, argmax
__global__ void k_final(const float* __restrict__ tok, const float* __restrict__ g,
                        const float* __restrict__ b, const float* __restrict__ w,
                        const float* __restrict__ bias, float* __restrict__ out) {
  __shared__ float red[512];
  int t = threadIdx.x;
  float xv = tok[t];
  red[t] = xv; __syncthreads();
  for (int st = 256; st > 0; st >>= 1) { if (t < st) red[t] += red[t + st]; __syncthreads(); }
  float mu = red[0] / 512.f; __syncthreads();
  float d = xv - mu;
  red[t] = d * d; __syncthreads();
  for (int st = 256; st > 0; st >>= 1) { if (t < st) red[t] += red[t + st]; __syncthreads(); }
  float rstd = rsqrtf(red[0] / 512.f + 1e-5f); __syncthreads();
  float ln = d * rstd * g[t] + b[t];
  red[t] = ln * w[t * 2 + 0]; __syncthreads();
  for (int st = 256; st > 0; st >>= 1) { if (t < st) red[t] += red[t + st]; __syncthreads(); }
  float l0 = red[0] + bias[0]; __syncthreads();
  red[t] = ln * w[t * 2 + 1]; __syncthreads();
  for (int st = 256; st > 0; st >>= 1) { if (t < st) red[t] += red[t + st]; __syncthreads(); }
  float l1 = red[0] + bias[1];
  if (t == 0) {
    float m = fmaxf(l0, l1);
    float e0 = __expf(l0 - m), e1 = __expf(l1 - m);
    float inv = 1.f / (e0 + e1);
    out[0] = l0; out[1] = l1; out[2] = e0 * inv; out[3] = e1 * inv;
    out[4] = (l1 > l0) ? 1.f : 0.f;
  }
}

// ---------------- Host orchestration ----------------
static inline unsigned cdiv(long long a, int b) { return (unsigned)((a + b - 1) / b); }

static void launch_gemm(hipStream_t s, const float* A, const float* B, float* C, const float* bias,
                        int M, int N, int K, int lda, int ldb, int ldc,
                        long long sA, long long sB, long long sC, int batch,
                        float alpha, int epi, int bnk)
{
  dim3 g(M / BM, N / BN, batch), blk(THREADS);
#define GO(E,T) gemm_kernel<E,T><<<g, blk, 0, s>>>(A, B, C, bias, M, N, K, lda, ldb, ldc, sA, sB, sC, alpha)
  if (bnk) { if (epi == 0) GO(0,1); else if (epi == 1) GO(1,1); else GO(2,1); }
  else     { if (epi == 0) GO(0,0); else if (epi == 1) GO(1,0); else GO(2,0); }
#undef GO
}

extern "C" void kernel_launch(void* const* d_in, const int* in_sizes, int n_in,
                              void* d_out, int out_size, void* d_ws, size_t ws_size,
                              hipStream_t stream)
{
  (void)in_sizes; (void)n_in; (void)out_size; (void)ws_size;
  const float* data  = (const float*)d_in[0];
  const float* fc1_w = (const float*)d_in[1];
  const float* fc1_b = (const float*)d_in[2];
  const float* cls   = (const float*)d_in[3];
  struct LW { const float *ng, *nb, *qkv, *ow, *ob, *rw; };
  LW L[2] = {
    { (const float*)d_in[4],  (const float*)d_in[5],  (const float*)d_in[6],
      (const float*)d_in[7],  (const float*)d_in[8],  (const float*)d_in[9] },
    { (const float*)d_in[10], (const float*)d_in[11], (const float*)d_in[12],
      (const float*)d_in[13], (const float*)d_in[14], (const float*)d_in[15] }
  };
  const float* p_w7 = (const float*)d_in[16]; const float* p_b7 = (const float*)d_in[17];
  const float* p_w5 = (const float*)d_in[18]; const float* p_b5 = (const float*)d_in[19];
  const float* p_w3 = (const float*)d_in[20]; const float* p_b3 = (const float*)d_in[21];
  const float* n_g  = (const float*)d_in[22]; const float* n_b  = (const float*)d_in[23];
  const float* fc2w = (const float*)d_in[24]; const float* fc2b = (const float*)d_in[25];

  float* W = (float*)d_ws;
  size_t off = 0;
  auto ALLOC = [&](size_t n) { float* p = W + off; off += n; return p; };
  float* tok  = ALLOC((size_t)NTOK * DMODEL);
  float* tmpf = ALLOC((size_t)16384 * DMODEL);         // fc1 out, then PPEG scratch
  float* yp   = ALLOC((size_t)NP * DMODEL);            // LN-padded input, reused as concat buf
  float* qkvb = ALLOC((size_t)NP * 3 * DMODEL);        // qkv, reused: attn_out + out-proj
  float* qb   = ALLOC((size_t)NHEAD * NP * DHEAD);
  float* kb   = ALLOC((size_t)NHEAD * NP * DHEAD);
  float* vb   = ALLOC((size_t)NHEAD * NP * DHEAD);
  float* s1   = ALLOC((size_t)NHEAD * NP * NLM);       // sim1 / a1
  float* s3   = ALLOC((size_t)NHEAD * NLM * NP);       // sim3 / a3
  float* ql   = ALLOC((size_t)NHEAD * NLM * DHEAD);
  float* kl   = ALLOC((size_t)NHEAD * NLM * DHEAD);
  float* a2   = ALLOC((size_t)NHEAD * NLM * NLM);
  float* z0   = ALLOC((size_t)NHEAD * NLM * NLM);
  float* z1   = ALLOC((size_t)NHEAD * NLM * NLM);
  float* xz   = ALLOC((size_t)NHEAD * NLM * NLM);
  float* wb   = ALLOC((size_t)NHEAD * NLM * NLM);
  float* tb   = ALLOC((size_t)NHEAD * NLM * NLM);
  float* a3vb = ALLOC((size_t)NHEAD * NLM * DHEAD);
  float* pvb  = ALLOC((size_t)NHEAD * NLM * DHEAD);

  float* attn_out = qkvb;                               // [8][NP][64]
  float* outp     = qkvb + (size_t)NP * DMODEL;         // [NP][512]
  float* cat      = yp;                                 // [NP][512]

  const long long MM = (long long)NLM * NLM;

  // fc1 + ReLU, then tokens = [cls; h]
  launch_gemm(stream, data, fc1_w, tmpf, fc1_b, 16384, DMODEL, DMODEL,
              DMODEL, DMODEL, DMODEL, 0, 0, 0, 1, 1.f, 2, 0);
  {
    long long n = (long long)NTOK * DMODEL;
    k_build_tokens<<<cdiv(n, THREADS), THREADS, 0, stream>>>(tmpf, cls, tok, n);
  }

  auto trans_layer = [&](const LW& lw) {
    long long n;
    n = (long long)PADR * DMODEL;
    k_fill_zero<<<cdiv(n, THREADS), THREADS, 0, stream>>>(yp, n);
    k_layernorm<<<NTOK, THREADS, 0, stream>>>(tok, lw.ng, lw.nb, yp + (size_t)PADR * DMODEL, DMODEL);

    launch_gemm(stream, yp, lw.qkv, qkvb, nullptr, NP, 3 * DMODEL, DMODEL,
                DMODEL, 3 * DMODEL, 3 * DMODEL, 0, 0, 0, 1, 1.f, 0, 0);
    n = (long long)NP * DMODEL;
    k_split_qkv<<<cdiv(n, THREADS), THREADS, 0, stream>>>(qkvb, qb, kb, vb, n);
    k_landmark<<<cdiv(NHEAD * NLM * DHEAD, THREADS), THREADS, 0, stream>>>(qb, ql);
    k_landmark<<<cdiv(NHEAD * NLM * DHEAD, THREADS), THREADS, 0, stream>>>(kb, kl);

    // sim1 = q @ k_l^T ; sim2 = q_l @ k_l^T ; sim3 = q_l @ k^T   (B^T form, BNK=1)
    launch_gemm(stream, qb, kl, s1, nullptr, NP, NLM, DHEAD, DHEAD, DHEAD, NLM,
                (long long)NP * DHEAD, (long long)NLM * DHEAD, (long long)NP * NLM, NHEAD, 1.f, 0, 1);
    launch_gemm(stream, ql, kl, a2, nullptr, NLM, NLM, DHEAD, DHEAD, DHEAD, NLM,
                (long long)NLM * DHEAD, (long long)NLM * DHEAD, MM, NHEAD, 1.f, 0, 1);
    launch_gemm(stream, ql, kb, s3, nullptr, NLM, NP, DHEAD, DHEAD, DHEAD, NP,
                (long long)NLM * DHEAD, (long long)NP * DHEAD, (long long)NLM * NP, NHEAD, 1.f, 0, 1);
    k_softmax<<<NHEAD * NP, THREADS, 0, stream>>>(s1, NLM);
    k_softmax<<<NHEAD * NLM, THREADS, 0, stream>>>(a2, NLM);
    k_softmax<<<NHEAD * NLM, THREADS, 0, stream>>>(s3, NP);

    // Moore-Penrose pinv of a2 (6 Newton-Schulz style iterations)
    k_pinv_init<<<NHEAD, NLM, 0, stream>>>(a2, z0);
    float* zc = z0; float* zn = z1;
    long long nmm = (long long)NHEAD * MM;
    for (int it = 0; it < 6; ++it) {
      launch_gemm(stream, a2, zc, xz, nullptr, NLM, NLM, NLM, NLM, NLM, NLM, MM, MM, MM, NHEAD, 1.f, 0, 0);
      k_id_minus<<<cdiv(nmm, THREADS), THREADS, 0, stream>>>(wb, xz, 7.f, nmm);
      launch_gemm(stream, xz, wb, tb, nullptr, NLM, NLM, NLM, NLM, NLM, NLM, MM, MM, MM, NHEAD, 1.f, 0, 0);
      k_id_minus<<<cdiv(nmm, THREADS), THREADS, 0, stream>>>(wb, tb, 15.f, nmm);
      launch_gemm(stream, xz, wb, tb, nullptr, NLM, NLM, NLM, NLM, NLM, NLM, MM, MM, MM, NHEAD, 1.f, 0, 0);
      k_id_minus<<<cdiv(nmm, THREADS), THREADS, 0, stream>>>(wb, tb, 13.f, nmm);
      launch_gemm(stream, zc, wb, zn, nullptr, NLM, NLM, NLM, NLM, NLM, NLM, MM, MM, MM, NHEAD, 0.25f, 0, 0);
      float* t2 = zc; zc = zn; zn = t2;
    }

    // out = a1 @ pinv @ (a3 @ v)
    launch_gemm(stream, s3, vb, a3vb, nullptr, NLM, DHEAD, NP, NP, DHEAD, DHEAD,
                (long long)NLM * NP, (long long)NP * DHEAD, (long long)NLM * DHEAD, NHEAD, 1.f, 0, 0);
    launch_gemm(stream, zc, a3vb, pvb, nullptr, NLM, DHEAD, NLM, NLM, DHEAD, DHEAD,
                MM, (long long)NLM * DHEAD, (long long)NLM * DHEAD, NHEAD, 1.f, 0, 0);
    launch_gemm(stream, s1, pvb, attn_out, nullptr, NP, DHEAD, NLM, NLM, DHEAD, DHEAD,
                (long long)NP * NLM, (long long)NLM * DHEAD, (long long)NP * DHEAD, NHEAD, 1.f, 0, 0);

    k_cat_conv<<<NP, THREADS, 0, stream>>>(attn_out, vb, lw.rw, cat);
    launch_gemm(stream, cat, lw.ow, outp, lw.ob, NP, DMODEL, DMODEL,
                DMODEL, DMODEL, DMODEL, 0, 0, 0, 1, 1.f, 1, 0);
    n = (long long)NTOK * DMODEL;
    k_res_add<<<cdiv(n, THREADS), THREADS, 0, stream>>>(tok, outp, n);
  };

  trans_layer(L[0]);
  k_ppeg<<<16384, THREADS, 0, stream>>>(tok, p_w7, p_b7, p_w5, p_b5, p_w3, p_b3, tmpf);
  k_copy<<<cdiv((long long)16384 * DMODEL, THREADS), THREADS, 0, stream>>>(tok + DMODEL, tmpf, (long long)16384 * DMODEL);
  trans_layer(L[1]);

  k_final<<<1, 512, 0, stream>>>(tok, n_g, n_b, fc2w, fc2b, (float*)d_out);
}
